// CategoricalDecoder_27187142983802
// MI455X (gfx1250) — matrix operand order repaired and checked by reading, not compiled, not run on hardware
//
#include <hip/hip_runtime.h>
#include <hip/hip_bf16.h>
#include <math.h>
#include <stdint.h>

// ---------------------------------------------------------------------------
// Problem constants (match reference setup_inputs)
// ---------------------------------------------------------------------------
#define B_   512
#define NZ_  16384
#define LAT_ 64
#define HID_ 512
#define D_   1536
#define K_   16

typedef __bf16 bf16_t;
typedef __attribute__((ext_vector_type(16))) __bf16 v16bf;
typedef __attribute__((ext_vector_type(8)))  __bf16 v8bf;
typedef __attribute__((ext_vector_type(8)))  float  v8f;
typedef uint32_t u32x4 __attribute__((ext_vector_type(4)));
typedef uint32_t u32x8 __attribute__((ext_vector_type(8)));

// ---------------------------------------------------------------------------
// WMMA helper: D = A(16x32 bf16) x B(32x16 bf16) + C(16x16 f32)
// ---------------------------------------------------------------------------
__device__ __forceinline__ v8f wmma_bf16(v16bf a, v16bf b, v8f c) {
    return __builtin_amdgcn_wmma_f32_16x16x32_bf16(
        /*neg_a=*/false, a, /*neg_b=*/false, b,
        /*c_mod=*/(short)0, c, /*reuse_a=*/false, /*reuse_b=*/false);
}

// A-matrix fragment (16x32, M x K) from row-major [.., ld] bf16.
// ISA layout: lane m=lane%16; elems 0..7 -> K = 8*(lane/16)+0..7,
//             elems 8..15 -> K = 16 + 8*(lane/16)+0..7.
__device__ __forceinline__ v16bf load_frag_a(const bf16_t* base, int ld,
                                             int m0, int k0) {
    const int lane = threadIdx.x & 31;
    const int m = m0 + (lane & 15);
    const int c = k0 + ((lane >> 4) << 3);
    const bf16_t* p = base + (size_t)m * ld + c;
    v8bf lo = *(const v8bf*)(p);
    v8bf hi = *(const v8bf*)(p + 16);
    v16bf a;
#pragma unroll
    for (int i = 0; i < 8; ++i) { a[i] = lo[i]; a[8 + i] = hi[i]; }
    return a;
}

// B-matrix fragment (32x16, K x N) where B[k][n] = T[n][k], T row-major [.., ld].
// ISA layout: lane n=lane%16; elems 0..7 -> K = 8*(lane/16)+0..7,
//             elems 8..15 -> K = 16 + 8*(lane/16)+0..7.
__device__ __forceinline__ v16bf load_frag_bt(const bf16_t* T, int ld,
                                              int n0, int k0) {
    const int lane = threadIdx.x & 31;
    const int n = n0 + (lane & 15);
    const int kb = k0 + ((lane >> 4) << 3);
    const bf16_t* p = T + (size_t)n * ld + kb;
    v8bf q0 = *(const v8bf*)(p);
    v8bf q1 = *(const v8bf*)(p + 16);
    v16bf b;
#pragma unroll
    for (int i = 0; i < 8; ++i) { b[i] = q0[i]; b[8 + i] = q1[i]; }
    return b;
}

// log_sigmoid(-L) = -softplus(L), numerically stable
__device__ __forceinline__ float log1m_sigmoid(float L) {
    return -(fmaxf(L, 0.0f) + log1pf(__expf(-fabsf(L))));
}

// ---------------------------------------------------------------------------
// TDM: async 2D tile load (bf16 elements) Global -> LDS, tracked by TENSORcnt.
// Descriptor per CDNA5 ISA.
//   tensor: tensor_h x tensor_w (row-major, row stride = stride_elems)
//   tile:   tile_h rows x tile_w cols starting at gsrc
// ---------------------------------------------------------------------------
__device__ __forceinline__ void tdm_load_tile_bf16(const bf16_t* gsrc,
                                                   bf16_t* ldst,
                                                   uint32_t tile_w,
                                                   uint32_t tile_h,
                                                   uint32_t tensor_w,
                                                   uint32_t tensor_h,
                                                   uint32_t stride_elems) {
    const uint64_t gaddr = (uint64_t)(uintptr_t)gsrc;
    const uint32_t laddr = (uint32_t)(uintptr_t)ldst;  // LDS byte offset

    u32x4 g0;
    g0[0] = 1u;                                           // count=1, no gather
    g0[1] = laddr;                                        // lds_addr
    g0[2] = (uint32_t)(gaddr & 0xFFFFFFFFu);              // global_addr[31:0]
    g0[3] = (uint32_t)((gaddr >> 32) & 0x01FFFFFFu)       // global_addr[56:32]
            | (2u << 30);                                 // type=2 ("image")

    u32x8 g1;
    g1[0] = 1u << 16;                    // data_size=1 (2 bytes/elem)
    g1[1] = (tensor_w & 0xFFFFu) << 16;  // tensor_dim0[15:0]
    g1[2] = ((tensor_w >> 16) & 0xFFFFu)          // tensor_dim0[31:16]
            | ((tensor_h & 0xFFFFu) << 16);       // tensor_dim1[15:0]
    g1[3] = ((tensor_h >> 16) & 0xFFFFu)          // tensor_dim1[31:16]
            | ((tile_w & 0xFFFFu) << 16);         // tile_dim0
    g1[4] = tile_h & 0xFFFFu;                     // tile_dim1 (tile_dim2=0)
    g1[5] = stride_elems;                         // tensor_dim0_stride[31:0]
    g1[6] = 0u;                                   // stride hi / dim1_stride lo
    g1[7] = 0u;

    asm volatile("tensor_load_to_lds %0, %1" :: "s"(g0), "s"(g1) : "memory");
}

// ---------------------------------------------------------------------------
// Prep kernels: f32 -> bf16 convert (and transpose variants)
// ---------------------------------------------------------------------------
__global__ void cvt_bf16_kernel(const float* __restrict__ src,
                                bf16_t* __restrict__ dst, int n) {
    int i = blockIdx.x * blockDim.x + threadIdx.x;
    if (i < n) dst[i] = (bf16_t)src[i];
}

// dst[c*rows + r] = src[r*cols + c]  (src: rows x cols f32 -> dst: cols x rows bf16)
__global__ void transpose_bf16_kernel(const float* __restrict__ src,
                                      bf16_t* __restrict__ dst,
                                      int rows, int cols) {
    int i = blockIdx.x * blockDim.x + threadIdx.x;
    if (i < rows * cols) {
        int r = i / cols, c = i % cols;
        dst[(size_t)c * rows + r] = (bf16_t)src[i];
    }
}

// ---------------------------------------------------------------------------
// Kernel 1: decode 16 bins per block.
//   H = relu(Zb @ W1 + b1)  (16x512, staged bf16 in LDS)
//   logits = H @ W2 + b2    (16x1536, stored bf16 to global, K-contiguous)
//   bias[bin] = sum_d log_sigmoid(-logit) + log_w[bin]
// Block: 128 threads (4 waves). Grid: NZ/16 = 1024.
// ---------------------------------------------------------------------------
__global__ void decode_kernel(const bf16_t* __restrict__ zb,     // [NZ, LAT]
                              const bf16_t* __restrict__ w1t,    // [HID, LAT]
                              const float*  __restrict__ b1,     // [HID]
                              const bf16_t* __restrict__ w2t,    // [D, HID]
                              const float*  __restrict__ b2,     // [D]
                              const float*  __restrict__ log_w,  // [NZ]
                              bf16_t* __restrict__ logitsb,      // [NZ, D]
                              float*  __restrict__ bias) {       // [NZ]
    const int bin0 = blockIdx.x * 16;
    const int wave = threadIdx.x >> 5;     // 0..3
    const int lane = threadIdx.x & 31;

    __shared__ bf16_t Hs[16 * HID_];       // 16 KB
    __shared__ float  rs[16];
    if (threadIdx.x < 16) rs[threadIdx.x] = 0.0f;

    // ---- GEMM1: H tiles (32 N-tiles over HID, K = LAT = 64) ----
    for (int t = wave; t < HID_ / 16; t += 4) {
        v8f acc = {};
        acc = wmma_bf16(load_frag_a(zb, LAT_, bin0, 0),
                        load_frag_bt(w1t, LAT_, t * 16, 0), acc);
        acc = wmma_bf16(load_frag_a(zb, LAT_, bin0, 32),
                        load_frag_bt(w1t, LAT_, t * 16, 32), acc);
        const int n  = t * 16 + (lane & 15);
        const int mb = (lane >> 4) << 3;
        const float bv = b1[n];
#pragma unroll
        for (int r = 0; r < 8; ++r) {
            float v = acc[r] + bv;
            Hs[(mb + r) * HID_ + n] = (bf16_t)(v > 0.0f ? v : 0.0f);
        }
    }
    __syncthreads();

    // ---- GEMM2: logits tiles (96 N-tiles over D, K = HID = 512) ----
    for (int t = wave; t < D_ / 16; t += 4) {
        v8f acc = {};
#pragma unroll 4
        for (int k0 = 0; k0 < HID_; k0 += 32) {
            v16bf a = load_frag_a((const bf16_t*)Hs, HID_, 0, k0);
            v16bf b = load_frag_bt(w2t, HID_, t * 16, k0);
            acc = wmma_bf16(a, b, acc);
        }
        const int n  = t * 16 + (lane & 15);
        const int mb = (lane >> 4) << 3;
        const float bv = b2[n];
#pragma unroll
        for (int r = 0; r < 8; ++r) {
            float L = acc[r] + bv;
            logitsb[(size_t)(bin0 + mb + r) * D_ + n] = (bf16_t)L;
            atomicAdd(&rs[mb + r], log1m_sigmoid(L));
        }
    }
    __syncthreads();

    if (threadIdx.x < 16) {
        int bin = bin0 + threadIdx.x;
        bias[bin] = rs[threadIdx.x] + log_w[bin];
    }
}

// ---------------------------------------------------------------------------
// Kernel 2: scores S[b,n] = xb[b,:] . logits[n,:] + bias[n]
// NT GEMM, K = D = 1536. Block: 256 threads (8 waves), computes 64x128 block.
// Grid: (B/64, NZ/128) = (8, 128).
// The streaming operand (logits, 48 MB bf16) is staged to LDS via the Tensor
// Data Mover (double-buffered, TENSORcnt-tracked); x (1.5 MB, reused 128x) is
// loaded straight from global (L0/L2 resident).
// ---------------------------------------------------------------------------
#define KCHUNK_ 256
#define NCHUNKS_ (D_ / KCHUNK_)   // 6

__global__ void score_kernel(const bf16_t* __restrict__ xb,       // [B, D]
                             const bf16_t* __restrict__ logitsb,  // [NZ, D]
                             const float*  __restrict__ bias,     // [NZ]
                             float* __restrict__ scores) {        // [B, NZ]
    const int m0   = blockIdx.x * 64;
    const int n0   = blockIdx.y * 128;
    const int wave = threadIdx.x >> 5;   // 0..7 -> one 16-bin N-tile each
    const int lane = threadIdx.x & 31;

    __shared__ bf16_t Bs[2][128 * KCHUNK_];   // 2 x 64 KB

    v8f acc[4] = {v8f{}, v8f{}, v8f{}, v8f{}};

    // prologue: kick off chunk 0
    if (wave == 0) {
        tdm_load_tile_bf16(logitsb + (size_t)n0 * D_ + 0, &Bs[0][0],
                           KCHUNK_, 128, D_, NZ_, D_);
    }

    for (int c = 0; c < NCHUNKS_; ++c) {
        if (wave == 0) {
            __builtin_amdgcn_s_wait_tensorcnt(0);   // chunk c landed in LDS
        }
        __syncthreads();                            // publish buffer c&1
        if (wave == 0 && c + 1 < NCHUNKS_) {
            // overwrites buf[(c+1)&1], which all waves finished reading at the
            // barrier above (it held chunk c-1)
            tdm_load_tile_bf16(logitsb + (size_t)n0 * D_ + (c + 1) * KCHUNK_,
                               &Bs[(c + 1) & 1][0], KCHUNK_, 128, D_, NZ_, D_);
        }
        const bf16_t* bbuf = &Bs[c & 1][0];
#pragma unroll 2
        for (int kk = 0; kk < KCHUNK_; kk += 32) {
            v16bf b = load_frag_bt(bbuf, KCHUNK_, wave * 16, kk);
            const int k0 = c * KCHUNK_ + kk;
#pragma unroll
            for (int i = 0; i < 4; ++i) {
                v16bf a = load_frag_a(xb, D_, m0 + i * 16, k0);
                acc[i] = wmma_bf16(a, b, acc[i]);
            }
        }
        __syncthreads();                            // done reading buf c&1
    }

    const int n  = n0 + wave * 16 + (lane & 15);
    const int mb = (lane >> 4) << 3;
    const float bv = bias[n];
#pragma unroll
    for (int i = 0; i < 4; ++i)
#pragma unroll
        for (int r = 0; r < 8; ++r)
            scores[(size_t)(m0 + i * 16 + mb + r) * NZ_ + n] = acc[i][r] + bv;
}

// ---------------------------------------------------------------------------
// Kernel 3: per-row top-16 + logsumexp(val - log_w[idx]) - log(16).
// One block (256 threads) per batch row.
// ---------------------------------------------------------------------------
__global__ void topk_lse_kernel(const float* __restrict__ scores,  // [B, NZ]
                                const float* __restrict__ log_w,   // [NZ]
                                float* __restrict__ out) {         // [B]
    const int b   = blockIdx.x;
    const int tid = threadIdx.x;
    const float NEG = -__builtin_inff();

    // local top-16 over a strided 64-element slice
    float vloc[K_];
    int   iloc[K_];
#pragma unroll
    for (int j = 0; j < K_; ++j) { vloc[j] = NEG; iloc[j] = 0; }
    for (int j = 0; j < NZ_ / 256; ++j) {
        int n = tid + j * 256;
        float v = scores[(size_t)b * NZ_ + n];
        if (v > vloc[K_ - 1]) {
            int p = K_ - 1;
            while (p > 0 && vloc[p - 1] < v) {
                vloc[p] = vloc[p - 1]; iloc[p] = iloc[p - 1]; --p;
            }
            vloc[p] = v; iloc[p] = n;
        }
    }

    __shared__ float cv[256 * K_];
    __shared__ int   ci[256 * K_];
    __shared__ int   head[256];
    __shared__ float bestv[256];
    __shared__ int   bestt[256];
    __shared__ float selv[K_];
    __shared__ int   seli[K_];

#pragma unroll
    for (int j = 0; j < K_; ++j) {
        cv[tid * K_ + j] = vloc[j];
        ci[tid * K_ + j] = iloc[j];
    }
    head[tid] = 0;
    __syncthreads();

    // 16 rounds of k-way merge: pick global max head, advance that list
    for (int r = 0; r < K_; ++r) {
        int h = head[tid];
        bestv[tid] = (h < K_) ? cv[tid * K_ + h] : NEG;
        bestt[tid] = tid;
        __syncthreads();
        for (int s = 128; s > 0; s >>= 1) {
            if (tid < s && bestv[tid + s] > bestv[tid]) {
                bestv[tid] = bestv[tid + s];
                bestt[tid] = bestt[tid + s];
            }
            __syncthreads();
        }
        if (tid == 0) {
            int t = bestt[0];
            selv[r] = bestv[0];
            seli[r] = ci[t * K_ + head[t]];
            head[t] += 1;
        }
        __syncthreads();
    }

    if (tid == 0) {
        float lp[K_], mx = NEG;
#pragma unroll
        for (int r = 0; r < K_; ++r) {
            lp[r] = selv[r] - log_w[seli[r]];
            mx = fmaxf(mx, lp[r]);
        }
        float s = 0.0f;
#pragma unroll
        for (int r = 0; r < K_; ++r) s += __expf(lp[r] - mx);
        out[b] = mx + __logf(s) - __logf((float)K_);
    }
}

// ---------------------------------------------------------------------------
// Host launcher
// ---------------------------------------------------------------------------
extern "C" void kernel_launch(void* const* d_in, const int* in_sizes, int n_in,
                              void* d_out, int out_size, void* d_ws, size_t ws_size,
                              hipStream_t stream) {
    (void)in_sizes; (void)n_in; (void)out_size; (void)ws_size;
    const float* x     = (const float*)d_in[0];  // [B, D]
    const float* log_w = (const float*)d_in[1];  // [NZ]
    const float* z     = (const float*)d_in[2];  // [NZ, LAT]
    const float* W1    = (const float*)d_in[3];  // [LAT, HID]
    const float* b1    = (const float*)d_in[4];  // [HID]
    const float* W2    = (const float*)d_in[5];  // [HID, D]
    const float* b2    = (const float*)d_in[6];  // [D]
    float* out = (float*)d_out;                  // [B]

    // workspace layout (256B aligned)
    char* ws = (char*)d_ws;
    size_t off = 0;
    auto carve = [&](size_t bytes) {
        char* p = ws + off;
        off = (off + bytes + 255) & ~(size_t)255;
        return p;
    };
    bf16_t* zb      = (bf16_t*)carve((size_t)NZ_ * LAT_ * sizeof(bf16_t));
    bf16_t* w1t     = (bf16_t*)carve((size_t)HID_ * LAT_ * sizeof(bf16_t));
    bf16_t* w2t     = (bf16_t*)carve((size_t)D_ * HID_ * sizeof(bf16_t));
    bf16_t* xb      = (bf16_t*)carve((size_t)B_ * D_ * sizeof(bf16_t));
    bf16_t* logitsb = (bf16_t*)carve((size_t)NZ_ * D_ * sizeof(bf16_t));
    float*  bias    = (float*) carve((size_t)NZ_ * sizeof(float));
    float*  scores  = (float*) carve((size_t)B_ * NZ_ * sizeof(float));

    // prep: convert / transpose to bf16
    {
        int n = NZ_ * LAT_;
        cvt_bf16_kernel<<<(n + 255) / 256, 256, 0, stream>>>(z, zb, n);
        n = B_ * D_;
        cvt_bf16_kernel<<<(n + 255) / 256, 256, 0, stream>>>(x, xb, n);
        n = LAT_ * HID_;
        transpose_bf16_kernel<<<(n + 255) / 256, 256, 0, stream>>>(W1, w1t, LAT_, HID_);
        n = HID_ * D_;
        transpose_bf16_kernel<<<(n + 255) / 256, 256, 0, stream>>>(W2, w2t, HID_, D_);
    }

    // decode: logits + per-bin bias
    decode_kernel<<<NZ_ / 16, 128, 0, stream>>>(zb, w1t, b1, w2t, b2, log_w,
                                                logitsb, bias);

    // score GEMM (TDM-staged B operand)
    score_kernel<<<dim3(B_ / 64, NZ_ / 128), 256, 0, stream>>>(xb, logitsb,
                                                               bias, scores);

    // top-k + logsumexp
    topk_lse_kernel<<<B_, 256, 0, stream>>>(scores, log_w, out);
}
